// CSLSTM_57526791963080
// MI455X (gfx1250) — matrix operand order repaired and checked by reading, not compiled, not run on hardware
//
#include <hip/hip_runtime.h>

// ---------------------------------------------------------------------------
// CDNA5 (gfx1250) CS-LSTM pipeline, wave32 + v_wmma_f32_16x16x32_f16.
// 5 kernels on one stream:
//   1) lstm_enc  (target,   128 WGs)   h_target -> ws
//   2) lstm_enc  (neighbors,1024 WGs)  h_neigh  -> ws
//   3) social_conv1 (sparse-grid implicit GEMM, 256 WGs) -> out1 f16 ws
//   4) social_conv2 + global maxpool (512 WGs)           -> pooled ws
//   5) decoder (fusion linear + 25-step WMMA LSTM + out linear) -> d_out
// ---------------------------------------------------------------------------

typedef _Float16 v8h  __attribute__((ext_vector_type(8)));
typedef _Float16 v16h __attribute__((ext_vector_type(16)));
typedef float    v8f  __attribute__((ext_vector_type(8)));

#define WSTR 520   // padded N-stride (halves) for 128x512 W in LDS
#define HSTR 136   // padded hidden stride (halves) for h tiles in LDS
#define O1STR 4616 // padded per-batch stride (halves) of conv1-activation tile

// A fragment (16x32 f16, MxK): lane<16 holds row M=lane, K in {base..base+7, base+16..base+23},
// lane>=16 same row, K shifted by 8 (ISA 7.12.2 16-bit A layout).
__device__ __forceinline__ v16h ldA(const _Float16* base, int rstride, int colbase, int lane) {
    const _Float16* p = base + (lane & 15) * rstride + colbase + (lane >> 4) * 8;
    v8h lo = *(const v8h*)(p);
    v8h hi = *(const v8h*)(p + 16);
    v16h r;
#pragma unroll
    for (int i = 0; i < 8; ++i) { r[i] = lo[i]; r[i + 8] = hi[i]; }
    return r;
}

// B fragment (32x16 f16, KxN): lane holds column N = nbase + (lane&15),
// K in {kc*32+8*(lane>>4) .. +7, +16 .. +23} (weights stored K-major, N contiguous in LDS).
__device__ __forceinline__ v16h ldB(const _Float16* base, int wstr, int kc, int nbase, int lane) {
    int k = kc * 32 + ((lane >> 4) << 3);
    const _Float16* p = base + k * wstr + nbase + (lane & 15);
    v16h r;
#pragma unroll
    for (int i = 0; i < 8; ++i) { r[i] = p[i * wstr]; r[i + 8] = p[(i + 16) * wstr]; }
    return r;
}

__device__ __forceinline__ v8f wmma16(v16h a, v16h b, v8f c) {
    return __builtin_amdgcn_wmma_f32_16x16x32_f16(false, a, false, b, (short)0, c, false, false);
}

__device__ __forceinline__ float sigm(float x) { return 1.0f / (1.0f + __expf(-x)); }
__device__ __forceinline__ float tanh_(float x) {
    float e = __expf(2.0f * x);
    return (e - 1.0f) / (e + 1.0f);
}

// ---------------------------------------------------------------------------
// LSTM encoder: 64 rows per WG, T=20, F=7, H=128. gates = x@WihT + h@WhhT + b.
// LDS: Whh^T f16 [128][WSTR], h f16 [64][HSTR], x f32 [64][140], Wih/bias f32.
// ---------------------------------------------------------------------------
__global__ void __launch_bounds__(256)
lstm_enc(const float* __restrict__ x, const float* __restrict__ w_ih,
         const float* __restrict__ w_hh, const float* __restrict__ b_ih,
         const float* __restrict__ b_hh, float* __restrict__ hout) {
    extern __shared__ char smem[];
    _Float16* Wt   = (_Float16*)smem;           // [128][WSTR]
    _Float16* h16  = Wt + 128 * WSTR;           // [64][HSTR]
    float*    xt   = (float*)(h16 + 64 * HSTR); // [64][140]
    float*    wih  = xt + 64 * 140;             // [512][8]
    float*    bias = wih + 512 * 8;             // [512]

    const int tid = threadIdx.x;
    const int lane = tid & 31;
    const int w = tid >> 5;
    const int rowbase = blockIdx.x * 64;
    const int kh = lane >> 4, ln15 = lane & 15;

    for (int i = tid; i < 512 * 128; i += 256) {       // transpose WhhT into LDS f16
        int n = i >> 7, k = i & 127;
        Wt[k * WSTR + n] = (_Float16)w_hh[i];
    }
    for (int i = tid; i < 64 * 140; i += 256) xt[i] = x[(size_t)rowbase * 140 + i];
    for (int i = tid; i < 512 * 7; i += 256) {
        int n = i / 7, f = i - n * 7;
        wih[n * 8 + f] = w_ih[i];
    }
    for (int i = tid; i < 512; i += 256) bias[i] = b_ih[i] + b_hh[i];
    for (int i = tid; i < 64 * HSTR; i += 256) h16[i] = (_Float16)0.0f;

    const int mbase = (w >> 1) * 16;
    v8f creg[4];
#pragma unroll
    for (int u = 0; u < 4; ++u)
#pragma unroll
        for (int r = 0; r < 8; ++r) creg[u][r] = 0.0f;

    for (int t = 0; t < 20; ++t) {
        __syncthreads();                    // h(t-1) complete
        v16h afr[4];
#pragma unroll
        for (int kc = 0; kc < 4; ++kc)
            afr[kc] = ldA(h16 + mbase * HSTR, HSTR, kc * 32, lane);
        __syncthreads();                    // all A reads done before h(t) writes

#pragma unroll
        for (int u = 0; u < 4; ++u) {
            const int nbh = ((w & 1) * 4 + u) * 16;   // hidden column tile
            v8f acc[4];
#pragma unroll
            for (int g = 0; g < 4; ++g) {             // i, f, g, o
                const int nb = g * 128 + nbh;
                v8f a = {0, 0, 0, 0, 0, 0, 0, 0};
#pragma unroll
                for (int kc = 0; kc < 4; ++kc)
                    a = wmma16(afr[kc], ldB(Wt, WSTR, kc, nb, lane), a);
                const int n = nb + ln15;
                const float bb = bias[n];
                float wv[7];
#pragma unroll
                for (int f = 0; f < 7; ++f) wv[f] = wih[n * 8 + f];
#pragma unroll
                for (int r = 0; r < 8; ++r) {         // per-step input contribution (F=7)
                    const float* xp = xt + (mbase + r + 8 * kh) * 140 + t * 7;
                    float s = bb;
#pragma unroll
                    for (int f = 0; f < 7; ++f) s += xp[f] * wv[f];
                    a[r] += s;
                }
                acc[g] = a;
            }
#pragma unroll
            for (int r = 0; r < 8; ++r) {
                float ig = sigm(acc[0][r]);
                float fg = sigm(acc[1][r]);
                float gg = tanh_(acc[2][r]);
                float og = sigm(acc[3][r]);
                float cc = fg * creg[u][r] + ig * gg;
                creg[u][r] = cc;
                float hh = og * tanh_(cc);
                h16[(mbase + r + 8 * kh) * HSTR + nbh + ln15] = (_Float16)hh;
            }
        }
    }
    __syncthreads();
    for (int i = tid; i < 64 * 128; i += 256) {
        int rr = i >> 7, hd = i & 127;
        hout[(size_t)(rowbase + rr) * 128 + hd] = (float)h16[rr * HSTR + hd];
    }
}

// ---------------------------------------------------------------------------
// Social conv1 as sparse implicit GEMM: only 9 occupied grid cells; for each
// output position p only the occupied cells inside its 3x3 window contribute
// a [32,128]x[128,64] WMMA GEMM with the matching 3x3 weight tap.
// ---------------------------------------------------------------------------
__global__ void __launch_bounds__(256)
social_conv1(const float* __restrict__ ht, const float* __restrict__ hn,
             const float* __restrict__ w1, const float* __restrict__ b1,
             _Float16* __restrict__ out1) {
    extern __shared__ char smem[];
    _Float16* hc  = (_Float16*)smem;        // [9*32][HSTR] occupied-cell h vectors
    _Float16* Wt1 = hc + 9 * 32 * HSTR;     // [9 taps][128 ic][72 pad oc]

    const int tid = threadIdx.x, lane = tid & 31, w = tid >> 5;
    const int bbase = blockIdx.x * 32;
    const int kh = lane >> 4, ln15 = lane & 15;

    for (int i = tid; i < 9 * 32 * 128; i += 256) {
        int cell = i / (32 * 128);
        int rem = i - cell * 32 * 128;
        int b = rem >> 7, k = rem & 127;
        float v = (cell == 0) ? ht[(size_t)(bbase + b) * 128 + k]
                              : hn[((size_t)(bbase + b) * 8 + (cell - 1)) * 128 + k];
        hc[(cell * 32 + b) * HSTR + k] = (_Float16)v;
    }
    for (int i = tid; i < 64 * 128 * 9; i += 256) {   // conv1_w [oc][ic][ky][kx] -> [tap][ic][oc]
        int oc = i / 1152, r = i - oc * 1152;
        int ic = r / 9, tap = r - ic * 9;
        Wt1[(tap * 128 + ic) * 72 + oc] = (_Float16)w1[i];
    }
    __syncthreads();

    const int occy[9] = {4, 0, 0, 0, 0, 0, 0, 0, 1};  // center + neighbors idx 1..8
    const int occx[9] = {4, 1, 2, 3, 4, 5, 6, 7, 0};

    for (int pp = 0; pp < 8; ++pp) {
        const int p = w * 8 + pp;                     // wave-uniform position
        const int py = p >> 3, px = p & 7;
#pragma unroll
        for (int mt = 0; mt < 2; ++mt) {
            v8f acc[4];
#pragma unroll
            for (int nt = 0; nt < 4; ++nt)
#pragma unroll
                for (int r = 0; r < 8; ++r) acc[nt][r] = 0.0f;
#pragma unroll
            for (int j = 0; j < 9; ++j) {
                int dy = occy[j] - py, dx = occx[j] - px;
                int act = (dy >= -1 && dy <= 1 && dx >= -1 && dx <= 1);
                act = __builtin_amdgcn_readfirstlane(act);  // scalar branch: keep EXEC all-ones for WMMA
                if (act) {
                    const int tap = (dy + 1) * 3 + (dx + 1);
                    const _Float16* Ab = hc + (j * 32 + mt * 16) * HSTR;
                    const _Float16* Bb = Wt1 + tap * 128 * 72;
                    v16h af[4];
#pragma unroll
                    for (int kc = 0; kc < 4; ++kc) af[kc] = ldA(Ab, HSTR, kc * 32, lane);
#pragma unroll
                    for (int nt = 0; nt < 4; ++nt)
#pragma unroll
                        for (int kc = 0; kc < 4; ++kc)
                            acc[nt] = wmma16(af[kc], ldB(Bb, 72, kc, nt * 16, lane), acc[nt]);
                }
            }
#pragma unroll
            for (int nt = 0; nt < 4; ++nt) {
                const int n = nt * 16 + ln15;
                const float bb = b1[n];
#pragma unroll
                for (int r = 0; r < 8; ++r) {
                    int br = bbase + mt * 16 + r + 8 * kh;
                    float v = acc[nt][r] + bb;
                    out1[((size_t)br * 64 + p) * 64 + n] = (_Float16)(v > 0.f ? v : 0.f);
                }
            }
        }
    }
}

// ---------------------------------------------------------------------------
// Conv2 (64->32ch, 3x3, SAME) + ReLU + global max pool, dense via WMMA.
// 16 batch rows per WG; per position p accumulate over valid taps, K=64 ic.
// ---------------------------------------------------------------------------
__global__ void __launch_bounds__(256)
social_conv2(const _Float16* __restrict__ out1, const float* __restrict__ w2,
             const float* __restrict__ b2, float* __restrict__ pooled) {
    extern __shared__ char smem[];
    _Float16* o1  = (_Float16*)smem;            // [16][64 pos][72 pad ic], batch stride O1STR
    _Float16* Wt2 = o1 + 16 * O1STR;            // [9 taps][64 ic][40 pad oc]
    float*    red = (float*)(Wt2 + 9 * 64 * 40);// [8 waves][16][32]

    const int tid = threadIdx.x, lane = tid & 31, w = tid >> 5;
    const int bbase = blockIdx.x * 16;
    const int kh = lane >> 4, ln15 = lane & 15;

    for (int i = tid; i < 16 * 64 * 64; i += 256) {
        int b = i >> 12, rem = i & 4095;
        int p = rem >> 6, ic = rem & 63;
        o1[b * O1STR + p * 72 + ic] = out1[((size_t)(bbase + b) * 64 + p) * 64 + ic];
    }
    for (int i = tid; i < 32 * 64 * 9; i += 256) {   // conv2_w [oc][ic][ky][kx] -> [tap][ic][oc]
        int oc = i / 576, r = i - oc * 576;
        int ic = r / 9, tap = r - ic * 9;
        Wt2[(tap * 64 + ic) * 40 + oc] = (_Float16)w2[i];
    }
    __syncthreads();

    v8f m0, m1;
#pragma unroll
    for (int r = 0; r < 8; ++r) { m0[r] = 0.0f; m1[r] = 0.0f; }  // ReLU outputs >= 0

    for (int pp = 0; pp < 8; ++pp) {
        const int p = w + pp * 8;        // wave-uniform
        const int py = p >> 3, px = p & 7;
        v8f a0, a1;
#pragma unroll
        for (int r = 0; r < 8; ++r) { a0[r] = 0.0f; a1[r] = 0.0f; }
#pragma unroll
        for (int ky = 0; ky < 3; ++ky) {
#pragma unroll
            for (int kx = 0; kx < 3; ++kx) {
                int qy = py + ky - 1, qx = px + kx - 1;
                int act = (qy >= 0 && qy < 8 && qx >= 0 && qx < 8);
                act = __builtin_amdgcn_readfirstlane(act);
                if (act) {
                    const int q = qy * 8 + qx, tap = ky * 3 + kx;
                    const _Float16* Ab = o1 + q * 72;
                    const _Float16* Bb = Wt2 + tap * 64 * 40;
#pragma unroll
                    for (int kc = 0; kc < 2; ++kc) {
                        v16h a = ldA(Ab, O1STR, kc * 32, lane);
                        a0 = wmma16(a, ldB(Bb, 40, kc, 0, lane), a0);
                        a1 = wmma16(a, ldB(Bb, 40, kc, 16, lane), a1);
                    }
                }
            }
        }
        const float bb0 = b2[ln15], bb1 = b2[16 + ln15];
#pragma unroll
        for (int r = 0; r < 8; ++r) {
            float v0 = a0[r] + bb0; v0 = v0 > 0.f ? v0 : 0.f;
            float v1 = a1[r] + bb1; v1 = v1 > 0.f ? v1 : 0.f;
            m0[r] = m0[r] > v0 ? m0[r] : v0;
            m1[r] = m1[r] > v1 ? m1[r] : v1;
        }
    }
#pragma unroll
    for (int r = 0; r < 8; ++r) {
        int m = r + 8 * kh;
        red[(w * 16 + m) * 32 + ln15] = m0[r];
        red[(w * 16 + m) * 32 + 16 + ln15] = m1[r];
    }
    __syncthreads();
    for (int o = tid; o < 16 * 32; o += 256) {
        float v = red[o];
#pragma unroll
        for (int ww = 1; ww < 8; ++ww) {
            float xx = red[ww * 512 + o];
            v = v > xx ? v : xx;
        }
        pooled[(size_t)(bbase + (o >> 5)) * 32 + (o & 31)] = v;
    }
}

// ---------------------------------------------------------------------------
// Decoder: fusion linear -> h0, then 25 autoregressive WMMA LSTM steps with
// input dim 2 (VALU) and H->2 output projection each step.
// ---------------------------------------------------------------------------
__global__ void __launch_bounds__(256)
decoder(const float* __restrict__ ht, const float* __restrict__ pooled,
        const float* __restrict__ fw, const float* __restrict__ fb,
        const float* __restrict__ dwih, const float* __restrict__ dwhh,
        const float* __restrict__ dbih, const float* __restrict__ dbhh,
        const float* __restrict__ ow, const float* __restrict__ ob,
        float* __restrict__ out) {
    extern __shared__ char smem[];
    _Float16* Wt   = (_Float16*)smem;            // [128][WSTR]
    _Float16* h16  = Wt + 128 * WSTR;            // [64][HSTR]
    float*    wih  = (float*)(h16 + 64 * HSTR);  // [512][2]
    float*    bias = wih + 1024;                 // [512]
    float*    outw = bias + 512;                 // [2][128]
    float*    outb = outw + 256;                 // [2]
    float*    inp  = outb + 2;                   // [64][2]

    const int tid = threadIdx.x, lane = tid & 31, w = tid >> 5;
    const int bbase = blockIdx.x * 64;
    const int kh = lane >> 4, ln15 = lane & 15;

    for (int i = tid; i < 512 * 128; i += 256) {
        int n = i >> 7, k = i & 127;
        Wt[k * WSTR + n] = (_Float16)dwhh[i];
    }
    for (int i = tid; i < 1024; i += 256) wih[i] = dwih[i];
    for (int i = tid; i < 512; i += 256) bias[i] = dbih[i] + dbhh[i];
    for (int i = tid; i < 256; i += 256) outw[i] = ow[i];
    if (tid < 2) outb[tid] = ob[tid];
    if (tid < 128) inp[tid] = 0.0f;

    // fused = tanh([h_target, pooled] @ fus_w.T + fus_b) -> initial hidden
    for (int o = tid; o < 64 * 128; o += 256) {
        int b = o >> 7, hh = o & 127;
        const float* fwr = fw + hh * 160;
        const float* hv = ht + (size_t)(bbase + b) * 128;
        const float* pv = pooled + (size_t)(bbase + b) * 32;
        float s = fb[hh];
        for (int k = 0; k < 128; ++k) s += hv[k] * fwr[k];
        for (int k = 0; k < 32; ++k) s += pv[k] * fwr[128 + k];
        h16[b * HSTR + hh] = (_Float16)tanh_(s);
    }

    const int mbase = (w >> 1) * 16;
    v8f creg[4];
#pragma unroll
    for (int u = 0; u < 4; ++u)
#pragma unroll
        for (int r = 0; r < 8; ++r) creg[u][r] = 0.0f;

    for (int t = 0; t < 25; ++t) {
        __syncthreads();
        v16h afr[4];
#pragma unroll
        for (int kc = 0; kc < 4; ++kc)
            afr[kc] = ldA(h16 + mbase * HSTR, HSTR, kc * 32, lane);
        __syncthreads();

#pragma unroll
        for (int u = 0; u < 4; ++u) {
            const int nbh = ((w & 1) * 4 + u) * 16;
            v8f acc[4];
#pragma unroll
            for (int g = 0; g < 4; ++g) {
                const int nb = g * 128 + nbh;
                v8f a = {0, 0, 0, 0, 0, 0, 0, 0};
#pragma unroll
                for (int kc = 0; kc < 4; ++kc)
                    a = wmma16(afr[kc], ldB(Wt, WSTR, kc, nb, lane), a);
                const int n = nb + ln15;
                const float bb = bias[n], w0 = wih[n * 2], w1 = wih[n * 2 + 1];
#pragma unroll
                for (int r = 0; r < 8; ++r) {
                    const int row = mbase + r + 8 * kh;
                    a[r] += bb + inp[row * 2] * w0 + inp[row * 2 + 1] * w1;
                }
                acc[g] = a;
            }
#pragma unroll
            for (int r = 0; r < 8; ++r) {
                float ig = sigm(acc[0][r]);
                float fg = sigm(acc[1][r]);
                float gg = tanh_(acc[2][r]);
                float og = sigm(acc[3][r]);
                float cc = fg * creg[u][r] + ig * gg;
                creg[u][r] = cc;
                float hh = og * tanh_(cc);
                h16[(mbase + r + 8 * kh) * HSTR + nbh + ln15] = (_Float16)hh;
            }
        }
        __syncthreads();     // h(t) ready for the output projection
        if (tid < 128) {
            int b = tid >> 1, j = tid & 1;
            const float* wj = outw + j * 128;
            float s = outb[j];
            for (int k = 0; k < 128; ++k) s += (float)h16[b * HSTR + k] * wj[k];
            out[((size_t)(bbase + b) * 25 + t) * 2 + j] = s;
            inp[b * 2 + j] = s;   // autoregressive feedback, consumed after next top barrier
        }
    }
}

// ---------------------------------------------------------------------------
extern "C" void kernel_launch(void* const* d_in, const int* in_sizes, int n_in,
                              void* d_out, int out_size, void* d_ws, size_t ws_size,
                              hipStream_t stream) {
    (void)in_sizes; (void)n_in; (void)out_size; (void)ws_size;

    const float* target   = (const float*)d_in[0];
    const float* neigh    = (const float*)d_in[1];
    // d_in[2] neigh_spatial, d_in[3] lane: unused by reference
    const float* enc_w_ih = (const float*)d_in[4];
    const float* enc_w_hh = (const float*)d_in[5];
    const float* enc_b_ih = (const float*)d_in[6];
    const float* enc_b_hh = (const float*)d_in[7];
    const float* nb_w_ih  = (const float*)d_in[8];
    const float* nb_w_hh  = (const float*)d_in[9];
    const float* nb_b_ih  = (const float*)d_in[10];
    const float* nb_b_hh  = (const float*)d_in[11];
    const float* conv1_w  = (const float*)d_in[12];
    const float* conv1_b  = (const float*)d_in[13];
    const float* conv2_w  = (const float*)d_in[14];
    const float* conv2_b  = (const float*)d_in[15];
    const float* fus_w    = (const float*)d_in[16];
    const float* fus_b    = (const float*)d_in[17];
    const float* dec_w_ih = (const float*)d_in[18];
    const float* dec_w_hh = (const float*)d_in[19];
    const float* dec_b_ih = (const float*)d_in[20];
    const float* dec_b_hh = (const float*)d_in[21];
    const float* out_w    = (const float*)d_in[22];
    const float* out_b    = (const float*)d_in[23];

    // workspace partition (~101 MiB total)
    char* ws = (char*)d_ws;
    float*    h_t    = (float*)ws;                                     //  8192*128 f32
    float*    h_n    = (float*)(ws + 4194304);                         // 65536*128 f32
    _Float16* o1     = (_Float16*)(ws + 4194304 + 33554432);           //  8192*64*64 f16
    float*    pooled = (float*)(ws + 4194304 + 33554432 + 67108864);   //  8192*32 f32

    constexpr size_t SMEM_ENC =
        (size_t)128 * WSTR * 2 + (size_t)64 * HSTR * 2 + ((size_t)64 * 140 + 512 * 8 + 512) * 4;
    constexpr size_t SMEM_C1 = (size_t)9 * 32 * HSTR * 2 + (size_t)9 * 128 * 72 * 2;
    constexpr size_t SMEM_C2 = (size_t)16 * O1STR * 2 + (size_t)9 * 64 * 40 * 2 + (size_t)8 * 512 * 4;
    constexpr size_t SMEM_DEC =
        (size_t)128 * WSTR * 2 + (size_t)64 * HSTR * 2 + ((size_t)1024 + 512 + 256 + 2 + 128) * 4;

    lstm_enc<<<128, 256, SMEM_ENC, stream>>>(target, enc_w_ih, enc_w_hh, enc_b_ih, enc_b_hh, h_t);
    lstm_enc<<<1024, 256, SMEM_ENC, stream>>>(neigh, nb_w_ih, nb_w_hh, nb_b_ih, nb_b_hh, h_n);
    social_conv1<<<256, 256, SMEM_C1, stream>>>(h_t, h_n, conv1_w, conv1_b, o1);
    social_conv2<<<512, 256, SMEM_C2, stream>>>(o1, conv2_w, conv2_b, pooled);
    decoder<<<128, 256, SMEM_DEC, stream>>>(h_t, pooled, fus_w, fus_b,
                                            dec_w_ih, dec_w_hh, dec_b_ih, dec_b_hh,
                                            out_w, out_b, (float*)d_out);
}
